// AdvancedFeatureTokenizer_61649960567110
// MI455X (gfx1250) — compile-verified, matrix operand order checked by reading, not run on hardware
//
#include <hip/hip_runtime.h>

// ---- problem constants (match reference) ----
constexpr int B  = 4096;
constexpr int F  = 128;
constexpr int D  = 256;
constexpr int H  = 64;
constexpr int NB = 32;
constexpr int T  = 8;
constexpr int L  = 8;
constexpr int D2 = D / 2;   // 128
constexpr int D4 = D / 4;   // 64
constexpr int FW_PER_F = 2 * 16 * 512;   // (s,nt) frags * 512 bf16 each = 16384 per feature

typedef __attribute__((ext_vector_type(16))) __bf16 v16bf;
typedef __attribute__((ext_vector_type(8)))  float  v8f;

__device__ __forceinline__ unsigned short f2bf_bits(float f) {
    union { float f; unsigned u; } v; v.f = f;
    unsigned u = v.u;
    return (unsigned short)((u + 0x7FFFu + ((u >> 16) & 1u)) >> 16);  // RNE
}

// Inverse of the A/B fragment K-mapping: K = j + 8*half + 8*(j>>3)
__device__ __forceinline__ void k_to_frag(int K, int& half, int& j) {
    int g = K >> 3;                 // 0..3
    half = g & 1;
    j = (K & 7) + 8 * (g >> 1);
}

// =====================================================================
// Kernel 1: per-feature fused weight  fused_w[f] = w2[f] @ proj_w[0:128,:]
//   stored bf16, pre-swizzled into WMMA B-fragment order with a
//   conflict-free LDS layout: frag(s,nt) = two 512B halves (p = j>>3),
//   within a half: lane*8 + (j&7)   (lane stride 16B -> all 64 banks).
//   Also pos_contrib[f,n] (pos_w + b2 + proj_b folded) and bin_contrib.
// grid = F + NB blocks, 256 threads
// =====================================================================
__global__ __launch_bounds__(256) void prep_kernel(
    const float* __restrict__ w2, const float* __restrict__ b2,
    const float* __restrict__ pos_w, const float* __restrict__ bin_emb_w,
    const float* __restrict__ proj_w, const float* __restrict__ proj_b,
    unsigned short* __restrict__ fw, float* __restrict__ pos_c,
    float* __restrict__ bin_c)
{
    const int t = threadIdx.x;
    const int blk = blockIdx.x;

    if (blk < F) {
        const int f = blk;
        __shared__ float sw2[H * D2];        // 32 KB
        __shared__ float sproj[16 * D];      // 16 KB: chunk of 16 rows of proj_w
        for (int i = t; i < H * D2; i += 256) sw2[i] = w2[(size_t)f * H * D2 + i];

        // per-thread 8x8 output tile of [64 x 256]
        const int th = t >> 5;          // 0..7  -> h base = th*8
        const int tn = t & 31;          // 0..31 -> n base = tn*8
        float acc[8][8];
        #pragma unroll
        for (int i = 0; i < 8; ++i)
            #pragma unroll
            for (int j = 0; j < 8; ++j) acc[i][j] = 0.f;

        for (int c0 = 0; c0 < D2; c0 += 16) {
            __syncthreads();
            for (int i = t; i < 16 * D; i += 256)
                sproj[i] = proj_w[(size_t)(c0 + (i >> 8)) * D + (i & 255)];
            __syncthreads();
            #pragma unroll
            for (int cc = 0; cc < 16; ++cc) {
                float a[8], b[8];
                #pragma unroll
                for (int i = 0; i < 8; ++i) a[i] = sw2[(th * 8 + i) * D2 + c0 + cc];
                #pragma unroll
                for (int j = 0; j < 8; ++j) b[j] = sproj[cc * D + tn * 8 + j];
                #pragma unroll
                for (int i = 0; i < 8; ++i)
                    #pragma unroll
                    for (int j = 0; j < 8; ++j) acc[i][j] += a[i] * b[j];
            }
        }
        // scatter into swizzled bf16 B-fragment order (bank-conflict-free halves)
        unsigned short* fwf = fw + (size_t)f * FW_PER_F;
        #pragma unroll
        for (int i = 0; i < 8; ++i) {
            const int h = th * 8 + i;
            const int s = h >> 5, K = h & 31;
            int half, j; k_to_frag(K, half, j);
            #pragma unroll
            for (int jj = 0; jj < 8; ++jj) {
                const int n = tn * 8 + jj;
                const int nt = n >> 4, nl = n & 15;
                const int lane = half * 16 + nl;
                const int frag = s * 16 + nt;
                fwf[frag * 512 + (j >> 3) * 256 + lane * 8 + (j & 7)] =
                    f2bf_bits(acc[i][jj]);
            }
        }
        // pos_contrib: one n per thread
        {
            const int n = t;
            float pc = proj_b[n];
            for (int c = 0; c < D4; ++c) pc += pos_w[f * D4 + c] * proj_w[(size_t)(D2 + c) * D + n];
            for (int c = 0; c < D2; ++c) pc += b2[f * D2 + c] * proj_w[(size_t)c * D + n];
            pos_c[f * D + n] = pc;
        }
    } else {
        const int bin = blk - F;
        const int n = t;
        float bc = 0.f;
        for (int c = 0; c < D4; ++c)
            bc += bin_emb_w[bin * D4 + c] * proj_w[(size_t)(D2 + D4 + c) * D + n];
        bin_c[bin * D + n] = bc;
    }
}

// =====================================================================
// Kernel 2: ODT -> odt_contrib[b,n] = (mean_t leaf mix) @ proj_w[384:448,:]
// grid = B/32, 256 threads, 32 batch rows per block
// =====================================================================
__global__ __launch_bounds__(256) void odt_kernel(
    const float* __restrict__ x, const float* __restrict__ sel_w,
    const float* __restrict__ sel_b, const float* __restrict__ leaf_w,
    const float* __restrict__ proj_w, float* __restrict__ odt_c)
{
    __shared__ float sx[32 * F];        // 16 KB
    __shared__ float sprobs[32 * T * L];// 8 KB
    __shared__ float sodt[32 * D4];     // 8 KB
    __shared__ float sproj[D4 * 64];    // 16 KB (64-col chunk of odt proj rows)

    const int t = threadIdx.x;
    const int base = blockIdx.x * 32;
    for (int i = t; i < 32 * F; i += 256) sx[i] = x[(size_t)base * F + i];
    __syncthreads();

    // phase 1: one (row, tree) per thread -> leaf probabilities
    {
        const int row = t >> 3, tree = t & 7;
        float dec[L];
        #pragma unroll
        for (int l = 0; l < L; ++l) dec[l] = sel_b[tree * L + l];
        const float* xr = sx + row * F;
        for (int f = 0; f < F; ++f) {
            const float xv = xr[f];
            const float* sw = sel_w + ((size_t)tree * F + f) * L;
            #pragma unroll
            for (int l = 0; l < L; ++l) dec[l] += xv * sw[l];
        }
        float ssum = 0.f;
        #pragma unroll
        for (int l = 0; l < L; ++l) { dec[l] = 1.f / (1.f + __expf(-dec[l])); ssum += dec[l]; }
        const float inv = 1.f / ssum;
        #pragma unroll
        for (int l = 0; l < L; ++l) sprobs[(row * T + tree) * L + l] = dec[l] * inv;
    }
    __syncthreads();

    // phase 2: (row, 8 c's) per thread, reduce trees, mean
    {
        const int row = t >> 3, cb = (t & 7) * 8;
        #pragma unroll
        for (int cc = 0; cc < 8; ++cc) {
            const int c = cb + cc;
            float a = 0.f;
            for (int tr = 0; tr < T; ++tr)
                #pragma unroll
                for (int l = 0; l < L; ++l)
                    a += sprobs[(row * T + tr) * L + l] * leaf_w[((size_t)tr * L + l) * D4 + c];
            sodt[row * D4 + c] = a * (1.f / (float)T);
        }
    }
    __syncthreads();

    // phase 3: odt_contrib = sodt @ proj_w[384:448,:], 64-col chunks in LDS
    for (int ch = 0; ch < D / 64; ++ch) {
        __syncthreads();
        for (int i = t; i < D4 * 64; i += 256)
            sproj[i] = proj_w[(size_t)(D + D4 + D4 + (i >> 6)) * D + ch * 64 + (i & 63)];
        __syncthreads();
        const int row = t >> 3, nb = (t & 7) * 8;
        const float* o = sodt + row * D4;
        #pragma unroll
        for (int jn = 0; jn < 8; ++jn) {
            const int nn = nb + jn;
            float a = 0.f;
            for (int c = 0; c < D4; ++c) a += o[c] * sproj[c * 64 + nn];
            odt_c[((size_t)(base + row)) * D + ch * 64 + nn] = a;
        }
    }
}

// =====================================================================
// Kernel 3 (main): per (128-row batch tile, feature f):
//   async-copy fused_w[f] (pre-swizzled bf16) into LDS, mish -> bf16
//   A-frags in regs, 16x16x32 bf16 WMMA over K=64, N=256; fused epilogue
//   (pos/bin/odt adds) with non-temporal stores (output is write-once).
// grid = (B/128, F), 256 threads = 8 waves; wave w owns 16-row M-tile.
// =====================================================================
__global__ __launch_bounds__(256) void main_kernel(
    const float* __restrict__ x, const float* __restrict__ w1,
    const float* __restrict__ b1, const float* __restrict__ bin_adjust,
    const unsigned short* __restrict__ fw, const float* __restrict__ pos_c,
    const float* __restrict__ bin_c, const float* __restrict__ odt_c,
    float* __restrict__ out)
{
    __shared__ __align__(16) unsigned short sfw[FW_PER_F];  // 32 KB
    __shared__ float sw1[H], sb1[H], sedges[NB + 1], sx[128];
    __shared__ int sbin[128];

    const int t = threadIdx.x;
    const int f = blockIdx.y;
    const int rowBase = blockIdx.x * 128;

    // ---- async global->LDS staging of fused_w[f]: 256 threads x 8 x 16B ----
    {
        const unsigned short* src = fw + (size_t)f * FW_PER_F;
        const unsigned goff = (unsigned)(t * 16);                 // bytes
        const unsigned lds0 = (unsigned)(uintptr_t)(&sfw[0]) + goff;
        #pragma unroll
        for (int i = 0; i < 8; ++i) {
            const unsigned lds_a = lds0 + (unsigned)(i * 4096);
            asm volatile("global_load_async_to_lds_b128 %0, %1, %2 offset:%3"
                         :: "v"(lds_a), "v"(goff), "s"(src), "n"(i * 4096)
                         : "memory");
        }
    }
    if (t < H) { sw1[t] = w1[f * H + t]; sb1[t] = b1[f * H + t]; }
    if (t >= 64 && t < 64 + NB + 1) {
        const int k = t - 64;
        sedges[k] = -4.f + 0.25f * (float)k + bin_adjust[f * (NB + 1) + k];
    }
    asm volatile("s_wait_asynccnt 0x0" ::: "memory");
    __syncthreads();

    // per-row x value + bucketize (searchsorted side='left', then clip)
    if (t < 128) {
        const float xv = x[(size_t)(rowBase + t) * F + f];
        sx[t] = xv;
        int i = 0;
        #pragma unroll
        for (int k = 0; k < NB + 1; ++k) i += (sedges[k] < xv) ? 1 : 0;
        i -= 1;
        if (i < 0) i = 0;
        if (i > NB - 1) i = NB - 1;
        sbin[t] = i;
    }
    __syncthreads();

    const int lane = t & 31, wave = t >> 5;
    const int half = lane >> 4, nl = lane & 15;

    // build bf16 A fragments directly in the ISA layout (16-bit A 16x32):
    // K(j,half) = j + 8*half + 8*(j>>3);  lane m = lane&15 holds row M=m.
    union AF { v16bf v; unsigned short u[16]; } af[2];
    {
        const float xv = sx[wave * 16 + nl];
        #pragma unroll
        for (int s = 0; s < 2; ++s) {
            #pragma unroll
            for (int j = 0; j < 16; ++j) {
                const int h = 32 * s + (j + 8 * half + 8 * (j >> 3));
                const float z = xv * sw1[h] + sb1[h];
                // mish(z) = z*tanh(softplus(z)) = z*(p^2-1)/(p^2+1), p = 1+e^z
                const float e = __expf(fminf(z, 15.f));
                const float p = 1.f + e;
                const float p2 = p * p;
                af[s].u[j] = f2bf_bits(z * (p2 - 1.f) / (p2 + 1.f));
            }
        }
    }

    // WMMA over 16 N-tiles x 2 K-steps, epilogue fused into the store
    #pragma unroll
    for (int nt = 0; nt < 16; ++nt) {
        v8f c = {0.f, 0.f, 0.f, 0.f, 0.f, 0.f, 0.f, 0.f};
        #pragma unroll
        for (int s = 0; s < 2; ++s) {
            // conflict-free B fragment: two 512B halves, lane stride 16B
            union BV { v16bf v; uint4 q[2]; } bv;
            const int fragbase = (s * 16 + nt) * 512;
            bv.q[0] = *reinterpret_cast<const uint4*>(sfw + fragbase + lane * 8);
            bv.q[1] = *reinterpret_cast<const uint4*>(sfw + fragbase + 256 + lane * 8);
            c = __builtin_amdgcn_wmma_f32_16x16x32_bf16(
                    false, af[s].v, false, bv.v, (short)0, c, false, false);
        }
        const int N = nt * 16 + nl;
        const float pc = pos_c[f * D + N];
        #pragma unroll
        for (int v = 0; v < 8; ++v) {
            const int lr = wave * 16 + v + 8 * half;   // C/D layout: M = v + 8*half
            const int b  = rowBase + lr;
            const float val = c[v] + pc
                            + bin_c[sbin[lr] * D + N]
                            + odt_c[(size_t)b * D + N];
            // output is write-once/never-read: non-temporal to preserve L2
            __builtin_nontemporal_store(val, &out[((size_t)b * F + f) * D + N]);
        }
    }
}

// =====================================================================
extern "C" void kernel_launch(void* const* d_in, const int* in_sizes, int n_in,
                              void* d_out, int out_size, void* d_ws, size_t ws_size,
                              hipStream_t stream) {
    const float* x          = (const float*)d_in[0];
    const float* w1         = (const float*)d_in[1];
    const float* b1         = (const float*)d_in[2];
    const float* w2         = (const float*)d_in[3];
    const float* b2         = (const float*)d_in[4];
    const float* pos_w      = (const float*)d_in[5];
    const float* bin_adjust = (const float*)d_in[6];
    const float* bin_emb_w  = (const float*)d_in[7];
    const float* sel_w      = (const float*)d_in[8];
    const float* sel_b      = (const float*)d_in[9];
    const float* leaf_w     = (const float*)d_in[10];
    const float* proj_w     = (const float*)d_in[11];
    const float* proj_b     = (const float*)d_in[12];

    char* ws = (char*)d_ws;
    unsigned short* fw = (unsigned short*)ws;                         // 4 MB bf16
    size_t off = (size_t)F * FW_PER_F * sizeof(unsigned short);
    float* pos_c = (float*)(ws + off);  off += (size_t)F * D * sizeof(float);
    float* bin_c = (float*)(ws + off);  off += (size_t)NB * D * sizeof(float);
    float* odt_c = (float*)(ws + off);  // 4 MB
    float* out = (float*)d_out;

    prep_kernel<<<F + NB, 256, 0, stream>>>(w2, b2, pos_w, bin_emb_w, proj_w, proj_b,
                                            fw, pos_c, bin_c);
    odt_kernel<<<B / 32, 256, 0, stream>>>(x, sel_w, sel_b, leaf_w, proj_w, odt_c);
    main_kernel<<<dim3(B / 128, F), 256, 0, stream>>>(x, w1, b1, bin_adjust,
                                                      fw, pos_c, bin_c, odt_c, out);
}
